// SelfAttention_70291434766657
// MI455X (gfx1250) — matrix operand order repaired and checked by reading, not compiled, not run on hardware
//
#include <hip/hip_runtime.h>
#include <hip/hip_bf16.h>

// ---------------------------------------------------------------------------
// Types for CDNA5 WMMA (wave32, 16x16x32 bf16 -> f32 accum)
// ---------------------------------------------------------------------------
typedef __attribute__((ext_vector_type(16))) __bf16 v16bf;
typedef __attribute__((ext_vector_type(8)))  float  v8f;
typedef __attribute__((ext_vector_type(4)))  unsigned int v4u;

union Frag {
    v16bf bf;
    v4u   u2[2];
    unsigned short us[16];
};

__device__ __forceinline__ unsigned short f2bf(float f) {
    unsigned int u = __float_as_uint(f);
    unsigned int r = u + 0x7FFFu + ((u >> 16) & 1u);  // round-to-nearest-even
    return (unsigned short)(r >> 16);
}
__device__ __forceinline__ float bf2f(unsigned short b) {
    return __uint_as_float(((unsigned int)b) << 16);
}

__device__ __forceinline__ v8f wmma_bf16(const Frag& a, const Frag& b, v8f c) {
    // (neg_a, A, neg_b, B, c_mod, C, reuse_a, reuse_b)
    return __builtin_amdgcn_wmma_f32_16x16x32_bf16(
        false, a.bf, false, b.bf, (short)0, c, false, false);
}

// CDNA5 async DMA: global memory -> LDS, tracked by ASYNCcnt (no VGPR staging).
// VDST VGPR carries the LDS byte offset (low 32 bits of the generic pointer);
// hardware adds the wave's LDS_BASE.  GV addressing: 64-bit address in VGPR pair.
__device__ __forceinline__ void async_load_b128(unsigned lds_off, const void* gptr) {
    asm volatile("global_load_async_to_lds_b128 %0, %1, off"
                 :: "v"(lds_off), "v"(gptr) : "memory");
}
__device__ __forceinline__ void wait_asynccnt0() {
    asm volatile("s_wait_asynccnt 0x0" ::: "memory");
}

#define DIMC     4096
#define SEQL     2048
#define NHEADS   32
#define HDIM     128
#define BATCHN   2

// ---------------------------------------------------------------------------
// fp32 -> bf16 conversion (vector x4)
// ---------------------------------------------------------------------------
__global__ void cvt_f32_bf16(const float* __restrict__ in,
                             unsigned short* __restrict__ out, int n) {
    int i = (blockIdx.x * blockDim.x + threadIdx.x) * 4;
    if (i + 3 < n) {
        float4 v = *(const float4*)(in + i);
        ushort4 o;
        o.x = f2bf(v.x); o.y = f2bf(v.y); o.z = f2bf(v.z); o.w = f2bf(v.w);
        *(ushort4*)(out + i) = o;
    }
}

// ---------------------------------------------------------------------------
// GEMM: C[M,N] = A[M,K](bf16) * W[N,K]^T(bf16) + bias[N]
// Block tile 128x128, 8 waves (4 along M x 2 along N), wave tile 32x64.
// Double-buffered LDS staging filled by GLOBAL_LOAD_ASYNC_TO_LDS_B128 DMA,
// overlapped with the WMMA stream; per-wave s_wait_asynccnt + barrier handoff.
// storeBf: 1 -> bf16 out, 0 -> f32 out
// ---------------------------------------------------------------------------
__global__ __launch_bounds__(256) void gemm_bf16(
    const unsigned short* __restrict__ A,
    const unsigned short* __restrict__ W,
    const float* __restrict__ bias,
    float* __restrict__ outF,
    unsigned short* __restrict__ outB,
    int M, int N, int K, int storeBf)
{
    __shared__ unsigned short sA[2][128 * 32];
    __shared__ unsigned short sW[2][128 * 32];

    const int tid = threadIdx.x;
    const int L   = tid & 31;
    const int w   = tid >> 5;
    const int wm  = w >> 1;        // 0..3
    const int wn  = w & 1;         // 0..1
    const int m0  = blockIdx.y * 128;
    const int n0  = blockIdx.x * 128;
    const int nk  = K >> 5;
    const int nn  = L & 15;
    const int hi  = L >> 4;

    v8f acc[2][4];
    v8f zero = {};
    #pragma unroll
    for (int i = 0; i < 2; ++i)
        #pragma unroll
        for (int j = 0; j < 4; ++j) acc[i][j] = zero;

    // Async DMA fill of one 128x32 A-tile + 128x32 W-tile (32B per thread each)
    auto loadTileAsync = [&](int buf, int kt) {
        int r = tid >> 1;
        int c = (tid & 1) * 16;
        unsigned la = (unsigned)(unsigned long long)&sA[buf][r * 32 + c];
        const unsigned short* ga = A + (size_t)(m0 + r) * K + kt * 32 + c;
        async_load_b128(la,      ga);
        async_load_b128(la + 16, ga + 8);
        unsigned lw = (unsigned)(unsigned long long)&sW[buf][r * 32 + c];
        const unsigned short* gw = W + (size_t)(n0 + r) * K + kt * 32 + c;
        async_load_b128(lw,      gw);
        async_load_b128(lw + 16, gw + 8);
    };

    loadTileAsync(0, 0);
    wait_asynccnt0();
    __syncthreads();

    for (int kt = 0; kt < nk; ++kt) {
        const int cur = kt & 1;
        if (kt + 1 < nk) loadTileAsync(cur ^ 1, kt + 1);

        // A fragments (lane m = L&15, K contiguous per half-wave)
        Frag af[2];
        #pragma unroll
        for (int i = 0; i < 2; ++i) {
            int ml = wm * 32 + i * 16 + nn;
            af[i].u2[0] = *reinterpret_cast<const v4u*>(&sA[cur][ml * 32 + hi * 16]);
            af[i].u2[1] = *reinterpret_cast<const v4u*>(&sA[cur][ml * 32 + hi * 16 + 8]);
        }
        // B fragments (lane n = L&15, K contiguous per half-wave) + WMMA
        #pragma unroll
        for (int j = 0; j < 4; ++j) {
            int nl = wn * 64 + j * 16 + nn;
            Frag wf;
            wf.u2[0] = *reinterpret_cast<const v4u*>(&sW[cur][nl * 32 + hi * 16]);
            wf.u2[1] = *reinterpret_cast<const v4u*>(&sW[cur][nl * 32 + hi * 16 + 8]);
            acc[0][j] = wmma_bf16(af[0], wf, acc[0][j]);
            acc[1][j] = wmma_bf16(af[1], wf, acc[1][j]);
        }
        // Own DMA done -> barrier makes whole next stage visible to all waves,
        // and guarantees everyone finished reading `cur` before it is refilled.
        wait_asynccnt0();
        __syncthreads();
    }

    // Epilogue: bias + store (C layout: lane holds rows r+8*hi, col n)
    #pragma unroll
    for (int j = 0; j < 4; ++j) {
        int nG = n0 + wn * 64 + j * 16 + nn;
        float bv = bias[nG];
        #pragma unroll
        for (int i = 0; i < 2; ++i) {
            #pragma unroll
            for (int r = 0; r < 8; ++r) {
                int mG = m0 + wm * 32 + i * 16 + r + 8 * hi;
                float v = acc[i][j][r] + bv;
                if (storeBf) outB[(size_t)mG * N + nG] = f2bf(v);
                else         outF[(size_t)mG * N + nG] = v;
            }
        }
    }
}

// ---------------------------------------------------------------------------
// RoPE applied in-place on bf16 Q and K ([B,S,H,D] layout, pairs (2i,2i+1))
// ---------------------------------------------------------------------------
__global__ void rope_kernel(unsigned short* __restrict__ Q,
                            unsigned short* __restrict__ Kq,
                            const float* __restrict__ fc,
                            const float* __restrict__ fs)
{
    int p = blockIdx.x * blockDim.x + threadIdx.x;   // B*S*H*64 = 8388608
    int i = p & 63;
    int h = (p >> 6) & 31;
    int s = (p >> 11) & 2047;
    int b = p >> 22;
    float c  = fc[s * 64 + i];
    float sn = fs[s * 64 + i];
    size_t base = ((size_t)(b * SEQL + s)) * DIMC + h * HDIM + 2 * i;

    float re = bf2f(Q[base]), im = bf2f(Q[base + 1]);
    Q[base]     = f2bf(re * c - im * sn);
    Q[base + 1] = f2bf(re * sn + im * c);

    re = bf2f(Kq[base]); im = bf2f(Kq[base + 1]);
    Kq[base]     = f2bf(re * c - im * sn);
    Kq[base + 1] = f2bf(re * sn + im * c);
}

// ---------------------------------------------------------------------------
// V: [B,S,H,D] -> Vt: [B,H,D,S]  (so P*V B-fragment is K(t)-contiguous)
// ---------------------------------------------------------------------------
__global__ void transpose_v(const unsigned short* __restrict__ V,
                            unsigned short* __restrict__ Vt)
{
    int idx = blockIdx.x * blockDim.x + threadIdx.x;  // 16777216
    int t = idx & 2047;
    int d = (idx >> 11) & 127;
    int h = (idx >> 18) & 31;
    int b = idx >> 23;
    Vt[idx] = V[((size_t)(b * SEQL + t)) * DIMC + h * HDIM + d];
}

// ---------------------------------------------------------------------------
// Causal flash attention. 1 wave = 16 query rows; block = 4 waves = 64 rows.
// Grid = B*H*(S/64) = 2048 blocks of 128 threads.
// Output stored with the reference's transpose(0,1,3,2).reshape index map:
//   y[b,h,s,d] -> Y[b*2048 + (h*128+d)/2][((h*128+d)&1)*2048 + s]
// ---------------------------------------------------------------------------
__global__ __launch_bounds__(128) void attn_kernel(
    const unsigned short* __restrict__ Q,
    const unsigned short* __restrict__ Kq,
    const unsigned short* __restrict__ Vt,
    unsigned short* __restrict__ Y)
{
    __shared__ unsigned short sP[4][16 * 32];

    const int tid = threadIdx.x;
    const int L   = tid & 31;
    const int w   = tid >> 5;
    const int bi  = blockIdx.x;
    const int q0  = (bi & 31) * 64 + w * 16;
    const int h   = (bi >> 5) & 31;
    const int b   = bi >> 10;
    const int nn  = L & 15;
    const int hi  = L >> 4;

    // Q fragments: 16 rows x 128 d, split into 4 K=32 chunks (A layout)
    Frag qf[4];
    {
        const unsigned short* qp =
            Q + ((size_t)(b * SEQL + q0 + nn)) * DIMC + h * HDIM;
        #pragma unroll
        for (int c = 0; c < 4; ++c) {
            qf[c].u2[0] = *(const v4u*)(qp + c * 32 + hi * 16);
            qf[c].u2[1] = *(const v4u*)(qp + c * 32 + hi * 16 + 8);
        }
    }

    v8f acc[8];
    v8f zero = {};
    #pragma unroll
    for (int j = 0; j < 8; ++j) acc[j] = zero;
    float rm[8], rl[8];
    #pragma unroll
    for (int r = 0; r < 8; ++r) { rm[r] = -3.0e38f; rl[r] = 0.0f; }

    const float scale = 0.08838834764831845f;   // 1/sqrt(128)
    const int nChunks = (q0 + 16 + 31) >> 5;

    for (int tc = 0; tc < nChunks; ++tc) {
        const int t0 = tc * 32;
        v8f s0 = zero, s1 = zero;
        const unsigned short* kp0 =
            Kq + ((size_t)(b * SEQL + t0 + nn)) * DIMC + h * HDIM;
        const unsigned short* kp1 = kp0 + (size_t)16 * DIMC;
        #pragma unroll
        for (int c = 0; c < 4; ++c) {
            Frag kf;
            kf.u2[0] = *(const v4u*)(kp0 + c * 32 + hi * 16);
            kf.u2[1] = *(const v4u*)(kp0 + c * 32 + hi * 16 + 8);
            s0 = wmma_bf16(qf[c], kf, s0);
            kf.u2[0] = *(const v4u*)(kp1 + c * 32 + hi * 16);
            kf.u2[1] = *(const v4u*)(kp1 + c * 32 + hi * 16 + 8);
            s1 = wmma_bf16(qf[c], kf, s1);
        }

        const bool needMask = (t0 + 31 > q0);
        #pragma unroll
        for (int r = 0; r < 8; ++r) {
            int mg = q0 + r + 8 * hi;
            float v0 = s0[r] * scale;
            float v1 = s1[r] * scale;
            if (needMask) {
                if (t0 + nn > mg)      v0 = -3.0e38f;
                if (t0 + 16 + nn > mg) v1 = -3.0e38f;
            }
            float mx = fmaxf(v0, v1);
            mx = fmaxf(mx, __shfl_xor(mx, 1, 32));
            mx = fmaxf(mx, __shfl_xor(mx, 2, 32));
            mx = fmaxf(mx, __shfl_xor(mx, 4, 32));
            mx = fmaxf(mx, __shfl_xor(mx, 8, 32));
            float mnew  = fmaxf(rm[r], mx);
            float alpha = __expf(rm[r] - mnew);
            float p0 = __expf(v0 - mnew);
            float p1 = __expf(v1 - mnew);
            float ls = p0 + p1;
            ls += __shfl_xor(ls, 1, 32);
            ls += __shfl_xor(ls, 2, 32);
            ls += __shfl_xor(ls, 4, 32);
            ls += __shfl_xor(ls, 8, 32);
            rl[r] = rl[r] * alpha + ls;
            rm[r] = mnew;
            #pragma unroll
            for (int j = 0; j < 8; ++j) acc[j][r] *= alpha;
            // stash P (C layout -> LDS row-major 16x32 tile)
            sP[w][(r + 8 * hi) * 32 + nn]      = f2bf(p0);
            sP[w][(r + 8 * hi) * 32 + 16 + nn] = f2bf(p1);
        }

        asm volatile("s_wait_dscnt 0" ::: "memory");

        // Re-read P as a WMMA A fragment (16x32)
        Frag pa;
        pa.u2[0] = *(const v4u*)&sP[w][nn * 32 + hi * 16];
        pa.u2[1] = *(const v4u*)&sP[w][nn * 32 + hi * 16 + 8];

        // P (16x32) x V (32x128) in 8 N-tiles of 16
        const unsigned short* vp =
            Vt + ((size_t)((b * NHEADS + h) * HDIM + nn)) * SEQL + t0 + hi * 16;
        #pragma unroll
        for (int j = 0; j < 8; ++j) {
            Frag vf;
            const unsigned short* vpj = vp + (size_t)j * 16 * SEQL;
            vf.u2[0] = *(const v4u*)(vpj);
            vf.u2[1] = *(const v4u*)(vpj + 8);
            acc[j] = wmma_bf16(pa, vf, acc[j]);
        }
    }

    // Normalize + scatter-store with the reference's permuted layout
    #pragma unroll
    for (int j = 0; j < 8; ++j) {
        int d  = j * 16 + nn;
        int hd = h * HDIM + d;
        size_t rowoff = ((size_t)(b * SEQL + (hd >> 1))) * DIMC + (hd & 1) * SEQL;
        #pragma unroll
        for (int r = 0; r < 8; ++r) {
            int s = q0 + r + 8 * hi;
            Y[rowoff + s] = f2bf(acc[j][r] / rl[r]);
        }
    }
}

// ---------------------------------------------------------------------------
// Host-side orchestration
// ---------------------------------------------------------------------------
extern "C" void kernel_launch(void* const* d_in, const int* in_sizes, int n_in,
                              void* d_out, int out_size, void* d_ws, size_t ws_size,
                              hipStream_t stream)
{
    const float* x    = (const float*)d_in[0];
    const float* wq_w = (const float*)d_in[1];
    const float* wq_b = (const float*)d_in[2];
    const float* wk_w = (const float*)d_in[3];
    const float* wk_b = (const float*)d_in[4];
    const float* wv_w = (const float*)d_in[5];
    const float* wv_b = (const float*)d_in[6];
    const float* wo_w = (const float*)d_in[7];
    const float* wo_b = (const float*)d_in[8];
    const float* fcos = (const float*)d_in[9];
    const float* fsin = (const float*)d_in[10];

    const size_t NE = (size_t)DIMC * DIMC;   // 16,777,216 elems (== B*S*DIM)
    unsigned short* ws  = (unsigned short*)d_ws;
    unsigned short* xb  = ws;
    unsigned short* wqb = xb  + NE;
    unsigned short* wkb = wqb + NE;
    unsigned short* wvb = wkb + NE;
    unsigned short* wob = wvb + NE;
    unsigned short* Qb  = wob + NE;
    unsigned short* Kb  = Qb  + NE;
    unsigned short* Vb  = Kb  + NE;
    unsigned short* Vt  = Vb  + NE;
    unsigned short* Yb  = Vt  + NE;

    const int cvtGrid = (int)(NE / (256 * 4));
    cvt_f32_bf16<<<cvtGrid, 256, 0, stream>>>(x,    xb,  (int)NE);
    cvt_f32_bf16<<<cvtGrid, 256, 0, stream>>>(wq_w, wqb, (int)NE);
    cvt_f32_bf16<<<cvtGrid, 256, 0, stream>>>(wk_w, wkb, (int)NE);
    cvt_f32_bf16<<<cvtGrid, 256, 0, stream>>>(wv_w, wvb, (int)NE);
    cvt_f32_bf16<<<cvtGrid, 256, 0, stream>>>(wo_w, wob, (int)NE);

    dim3 gG(32, 32), gB(256);
    gemm_bf16<<<gG, gB, 0, stream>>>(xb, wqb, wq_b, nullptr, Qb, DIMC, DIMC, DIMC, 1);
    gemm_bf16<<<gG, gB, 0, stream>>>(xb, wkb, wk_b, nullptr, Kb, DIMC, DIMC, DIMC, 1);
    gemm_bf16<<<gG, gB, 0, stream>>>(xb, wvb, wv_b, nullptr, Vb, DIMC, DIMC, DIMC, 1);

    rope_kernel<<<8388608 / 256, 256, 0, stream>>>(Qb, Kb, fcos, fsin);
    transpose_v<<<(int)(NE / 256), 256, 0, stream>>>(Vb, Vt);

    attn_kernel<<<2048, 128, 0, stream>>>(Qb, Kb, Vt, Yb);

    gemm_bf16<<<gG, gB, 0, stream>>>(Yb, wob, wo_b, (float*)d_out, nullptr,
                                     DIMC, DIMC, DIMC, 0);
}
